// MultiHeadSelfAttention_9646496547646
// MI455X (gfx1250) — matrix operand order repaired and checked by reading, not compiled, not run on hardware
//
#include <hip/hip_runtime.h>

// ---------------------------------------------------------------------------
// MI455X (gfx1250) multi-head self-attention, bf16 WMMA, f32 accumulate.
// B=8, N=1024, E=1024, H=16, D=64. SCALE = 1/8 folded into Q.
// ---------------------------------------------------------------------------

typedef __attribute__((ext_vector_type(16))) __bf16 v16bf;
typedef __attribute__((ext_vector_type(8)))  __bf16 v8bf;
typedef __attribute__((ext_vector_type(2)))  __bf16 v2bf;
typedef __attribute__((ext_vector_type(8)))  float  v8f;

// ---- fp32 -> bf16 conversion (lowers to v_cvt_pk_bf16_f32) -----------------
__device__ __forceinline__ unsigned pkbf(float a, float b) {
  v2bf v;
  v[0] = (__bf16)a;
  v[1] = (__bf16)b;
  return __builtin_bit_cast(unsigned, v);
}
__device__ __forceinline__ __bf16 f2bf(float f) { return (__bf16)f; }

__device__ __forceinline__ v8f v8f_zero() {
  v8f z;
#pragma unroll
  for (int i = 0; i < 8; ++i) z[i] = 0.f;
  return z;
}

__device__ __forceinline__ v16bf pack16(v8bf lo, v8bf hi) {
  v16bf r;
#pragma unroll
  for (int i = 0; i < 8; ++i) { r[i] = lo[i]; r[i + 8] = hi[i]; }
  return r;
}

// A-fragment (16x32 bf16): lane holds row (lane&15); element e<8 -> K=k0+e,
// e>=8 -> K=k0+16+(e-8). Caller passes p = &row[k0] with k0 = base + (lane>>4)*8.
__device__ __forceinline__ v16bf load_a_frag(const __bf16* p) {
  v8bf lo = *(const v8bf*)(p);
  v8bf hi = *(const v8bf*)(p + 16);
  return pack16(lo, hi);
}
// B-fragment (32x16 bf16): lane holds column (lane&15); element e<8 -> K=k0+e,
// e>=8 -> K=k0+16+(e-8). Caller passes p = &col_k0 with k0 = base + (lane>>4)*8.
__device__ __forceinline__ v16bf load_b_frag(const __bf16* p) {
  v8bf lo = *(const v8bf*)(p);
  v8bf hi = *(const v8bf*)(p + 16);
  return pack16(lo, hi);
}

__device__ __forceinline__ v8f wmma_bf16(v16bf a, v16bf b, v8f c) {
  return __builtin_amdgcn_wmma_f32_16x16x32_bf16(false, a, false, b,
                                                 (short)0, c, false, false);
}

#define EMB   1024
#define NSEQ  1024
#define NB    8
#define NH    16
#define HD    64
#define QSCALE 0.125f
#define AP    40   // padded LDS row stride (bf16) for 32-wide K tiles

// ---------------------------------------------------------------------------
// Kernel 1: QKV projection. out[f = 0..3071] = x @ w_qkv^T + b_qkv.
// Writes Q*SCALE -> qb[B,H,N,D], K -> kb[B,H,N,D], V -> vt[B,H,D,N] (bf16).
// Block: 128 threads (4 waves), tile 128(M) x 128(F), wave tile 64x64.
// ---------------------------------------------------------------------------
__global__ __launch_bounds__(128)
void qkv_kernel(const float* __restrict__ x, const float* __restrict__ w,
                const float* __restrict__ bias,
                __bf16* __restrict__ qb, __bf16* __restrict__ kb,
                __bf16* __restrict__ vt) {
  __shared__ __bf16 As[128][AP];
  __shared__ __bf16 Bs[128][AP];

  const int t    = threadIdx.x;
  const int lane = t & 31;
  const int wave = t >> 5;
  const int hi8  = lane >> 4;     // which 16-lane half
  const int ln   = lane & 15;
  const int mBlock = blockIdx.x * 128;
  const int fBlock = blockIdx.y * 128;
  const int wm = (wave >> 1) * 64;   // wave row offset in block tile
  const int wf = (wave & 1) * 64;    // wave col offset in block tile

  v8f acc[4][4];
#pragma unroll
  for (int i = 0; i < 4; ++i)
#pragma unroll
    for (int j = 0; j < 4; ++j) acc[i][j] = v8f_zero();

  for (int k0 = 0; k0 < EMB; k0 += 32) {
    // Pass 1: issue ALL global loads (16 in flight) before any conversion.
    float4 av[8], bv[8];
#pragma unroll
    for (int i = 0; i < 8; ++i) {
      const int idx = i * 128 + t;          // 0..1023
      const int row = idx >> 3;             // 0..127
      const int kq  = (idx & 7) * 4;        // 0..28
      av[i] = *(const float4*)(x + (size_t)(mBlock + row) * EMB + k0 + kq);
      bv[i] = *(const float4*)(w + (size_t)(fBlock + row) * EMB + k0 + kq);
    }
    // Pass 2: convert (v_cvt_pk_bf16_f32) and store to LDS.
#pragma unroll
    for (int i = 0; i < 8; ++i) {
      const int idx = i * 128 + t;
      const int row = idx >> 3;
      const int kq  = (idx & 7) * 4;
      uint2 ua;
      ua.x = pkbf(av[i].x, av[i].y);
      ua.y = pkbf(av[i].z, av[i].w);
      *(uint2*)&As[row][kq] = ua;
      uint2 ub;
      ub.x = pkbf(bv[i].x, bv[i].y);
      ub.y = pkbf(bv[i].z, bv[i].w);
      *(uint2*)&Bs[row][kq] = ub;
    }
    if (k0 + 32 < EMB) {   // lowers to global_prefetch_b8
      __builtin_prefetch(x + (size_t)(mBlock + t) * EMB + k0 + 32, 0, 1);
      __builtin_prefetch(w + (size_t)(fBlock + t) * EMB + k0 + 32, 0, 1);
    }
    __syncthreads();

    v16bf af[4];
#pragma unroll
    for (int i = 0; i < 4; ++i)
      af[i] = load_a_frag(&As[wm + i * 16 + ln][hi8 * 8]);
#pragma unroll
    for (int j = 0; j < 4; ++j) {
      const v16bf bj = load_b_frag(&Bs[wf + j * 16 + ln][hi8 * 8]);
#pragma unroll
      for (int i = 0; i < 4; ++i)
        acc[i][j] = wmma_bf16(af[i], bj, acc[i][j]);
    }
    __syncthreads();
  }

  // Epilogue: bias, scale Q, scatter to Q/K/V^T bf16 tensors.
  const int mW = mBlock + wm, fW = fBlock + wf;
#pragma unroll
  for (int i = 0; i < 4; ++i) {
#pragma unroll
    for (int j = 0; j < 4; ++j) {
      const int f   = fW + j * 16 + ln;
      const float bia = bias[f];
      const int seg = f >> 10;          // 0=Q 1=K 2=V (uniform per 16-col tile)
      const int fin = f & (EMB - 1);
      const int h = fin >> 6, d = fin & (HD - 1);
#pragma unroll
      for (int r = 0; r < 8; ++r) {
        const int m = mW + i * 16 + r + hi8 * 8;
        const int b = m >> 10, n = m & (NSEQ - 1);
        const size_t bh = (size_t)(b * NH + h);
        const float val = acc[i][j][r] + bia;
        if (seg == 0)      qb[(bh * NSEQ + n) * HD + d] = f2bf(val * QSCALE);
        else if (seg == 1) kb[(bh * NSEQ + n) * HD + d] = f2bf(val);
        else               vt[(bh * HD + d) * NSEQ + n] = f2bf(val);
      }
    }
  }
}

// ---------------------------------------------------------------------------
// Kernel 2: flash attention. Block = 4 waves x 32 query rows = 128 q rows.
// grid = (N/128, B*H). K/V tensors are L2-resident (32 MB total).
// ---------------------------------------------------------------------------
__global__ __launch_bounds__(128)
void attn_kernel(const __bf16* __restrict__ qb, const __bf16* __restrict__ kb,
                 const __bf16* __restrict__ vt, __bf16* __restrict__ attn) {
  __shared__ __bf16 P[4][32][72];   // per-wave P transpose buffer (C->A layout)

  const int t    = threadIdx.x;
  const int lane = t & 31;
  const int wave = t >> 5;
  const int hi8  = lane >> 4;
  const int ln   = lane & 15;
  const int bh   = blockIdx.y;
  const int qbase = blockIdx.x * 128 + wave * 32;

  const __bf16* Q = qb + (size_t)bh * NSEQ * HD;
  const __bf16* K = kb + (size_t)bh * NSEQ * HD;
  const __bf16* V = vt + (size_t)bh * HD * NSEQ;

  // Q fragments, reused across all KV tiles. aq[rowgroup][kstep]
  v16bf aq[2][2];
#pragma unroll
  for (int i = 0; i < 2; ++i)
#pragma unroll
    for (int s = 0; s < 2; ++s)
      aq[i][s] = load_a_frag(Q + (size_t)(qbase + i * 16 + ln) * HD + s * 32 + hi8 * 8);

  v8f  O[2][4];
  float rm[2][8], rs[2][8];
#pragma unroll
  for (int i = 0; i < 2; ++i) {
#pragma unroll
    for (int j = 0; j < 4; ++j) O[i][j] = v8f_zero();
#pragma unroll
    for (int r = 0; r < 8; ++r) { rm[i][r] = -1e30f; rs[i][r] = 0.f; }
  }

  for (int kv = 0; kv < NSEQ; kv += 64) {
    // S = Q * K^T  (32 x 64)
    v8f S[2][4];
#pragma unroll
    for (int i = 0; i < 2; ++i)
#pragma unroll
      for (int j = 0; j < 4; ++j) S[i][j] = v8f_zero();

#pragma unroll
    for (int j = 0; j < 4; ++j) {
      const __bf16* kp = K + (size_t)(kv + j * 16 + ln) * HD + hi8 * 8;
      const v16bf b0 = load_b_frag(kp);        // d = 0..31 half
      const v16bf b1 = load_b_frag(kp + 32);   // d = 32..63 half
#pragma unroll
      for (int i = 0; i < 2; ++i) {
        S[i][j] = wmma_bf16(aq[i][0], b0, S[i][j]);
        S[i][j] = wmma_bf16(aq[i][1], b1, S[i][j]);
      }
    }

    // Online softmax. Lanes 0-15 own rows r, lanes 16-31 own rows 8+r;
    // 16-lane xor reductions stay inside each half.
#pragma unroll
    for (int i = 0; i < 2; ++i) {
#pragma unroll
      for (int r = 0; r < 8; ++r) {
        float mx = rm[i][r];
#pragma unroll
        for (int j = 0; j < 4; ++j) mx = fmaxf(mx, S[i][j][r]);
#pragma unroll
        for (int off = 1; off < 16; off <<= 1)
          mx = fmaxf(mx, __shfl_xor(mx, off, 32));
        const float alpha = __expf(rm[i][r] - mx);
        rm[i][r] = mx;
        float ls = 0.f;
#pragma unroll
        for (int j = 0; j < 4; ++j) {
          const float p = __expf(S[i][j][r] - mx);
          ls += p;
          S[i][j][r] = p;            // S now holds P
        }
#pragma unroll
        for (int off = 1; off < 16; off <<= 1)
          ls += __shfl_xor(ls, off, 32);
        rs[i][r] = rs[i][r] * alpha + ls;
#pragma unroll
        for (int dj = 0; dj < 4; ++dj) O[i][dj][r] *= alpha;
      }
    }

    // P (C layout) -> LDS -> A layout. Per-wave region, DS ops are in-order.
#pragma unroll
    for (int i = 0; i < 2; ++i)
#pragma unroll
      for (int j = 0; j < 4; ++j)
#pragma unroll
        for (int r = 0; r < 8; ++r)
          P[wave][i * 16 + r + hi8 * 8][j * 16 + ln] = f2bf(S[i][j][r]);

    // O += P * V  (V^T rows give contiguous B fragments)
#pragma unroll
    for (int dj = 0; dj < 4; ++dj) {
      const __bf16* vp = V + (size_t)(dj * 16 + ln) * NSEQ + kv + hi8 * 8;
      const v16bf b0 = load_b_frag(vp);        // keys kv+0..31 half
      const v16bf b1 = load_b_frag(vp + 32);   // keys kv+32..63 half
#pragma unroll
      for (int i = 0; i < 2; ++i) {
        const v16bf a0 = load_a_frag(&P[wave][i * 16 + ln][hi8 * 8]);
        const v16bf a1 = load_a_frag(&P[wave][i * 16 + ln][32 + hi8 * 8]);
        O[i][dj] = wmma_bf16(a0, b0, O[i][dj]);
        O[i][dj] = wmma_bf16(a1, b1, O[i][dj]);
      }
    }
  }

  // Normalize and store to attn[B, N, E] (bf16), col = h*64 + d.
  const int b = bh >> 4, h = bh & (NH - 1);
#pragma unroll
  for (int i = 0; i < 2; ++i)
#pragma unroll
    for (int dj = 0; dj < 4; ++dj)
#pragma unroll
      for (int r = 0; r < 8; ++r) {
        const int n = qbase + i * 16 + r + hi8 * 8;
        const int c = h * HD + dj * 16 + ln;
        attn[((size_t)b * NSEQ + n) * EMB + c] = f2bf(O[i][dj][r] / rs[i][r]);
      }
}

// ---------------------------------------------------------------------------
// Kernel 3: output projection. out = attn @ w_proj^T + b_proj (fp32 out).
// ---------------------------------------------------------------------------
__global__ __launch_bounds__(128)
void proj_kernel(const __bf16* __restrict__ a, const float* __restrict__ w,
                 const float* __restrict__ bias, float* __restrict__ out) {
  __shared__ __bf16 As[128][AP];
  __shared__ __bf16 Bs[128][AP];

  const int t    = threadIdx.x;
  const int lane = t & 31;
  const int wave = t >> 5;
  const int hi8  = lane >> 4;
  const int ln   = lane & 15;
  const int mBlock = blockIdx.x * 128;
  const int fBlock = blockIdx.y * 128;
  const int wm = (wave >> 1) * 64;
  const int wf = (wave & 1) * 64;

  v8f acc[4][4];
#pragma unroll
  for (int i = 0; i < 4; ++i)
#pragma unroll
    for (int j = 0; j < 4; ++j) acc[i][j] = v8f_zero();

  for (int k0 = 0; k0 < EMB; k0 += 32) {
    // Pass 1: issue all global loads before converting/storing.
    uint4  avu[4];
    float4 bv[8];
#pragma unroll
    for (int i = 0; i < 4; ++i) {
      const int idx = i * 128 + t;        // 0..511
      const int row = idx >> 2;           // 0..127
      const int q   = (idx & 3) * 8;      // 0..24
      avu[i] = *(const uint4*)(a + (size_t)(mBlock + row) * EMB + k0 + q);
    }
#pragma unroll
    for (int i = 0; i < 8; ++i) {
      const int idx = i * 128 + t;
      const int row = idx >> 3;
      const int kq  = (idx & 7) * 4;
      bv[i] = *(const float4*)(w + (size_t)(fBlock + row) * EMB + k0 + kq);
    }
    // Pass 2: store A (raw bf16 copy) and B (converted) to LDS.
#pragma unroll
    for (int i = 0; i < 4; ++i) {
      const int idx = i * 128 + t;
      const int row = idx >> 2;
      const int q   = (idx & 3) * 8;
      *(uint4*)&As[row][q] = avu[i];
    }
#pragma unroll
    for (int i = 0; i < 8; ++i) {
      const int idx = i * 128 + t;
      const int row = idx >> 3;
      const int kq  = (idx & 7) * 4;
      uint2 ub;
      ub.x = pkbf(bv[i].x, bv[i].y);
      ub.y = pkbf(bv[i].z, bv[i].w);
      *(uint2*)&Bs[row][kq] = ub;
    }
    if (k0 + 32 < EMB)
      __builtin_prefetch(w + (size_t)(fBlock + t) * EMB + k0 + 32, 0, 1);
    __syncthreads();

    v16bf af[4];
#pragma unroll
    for (int i = 0; i < 4; ++i)
      af[i] = load_a_frag(&As[wm + i * 16 + ln][hi8 * 8]);
#pragma unroll
    for (int j = 0; j < 4; ++j) {
      const v16bf bj = load_b_frag(&Bs[wf + j * 16 + ln][hi8 * 8]);
#pragma unroll
      for (int i = 0; i < 4; ++i)
        acc[i][j] = wmma_bf16(af[i], bj, acc[i][j]);
    }
    __syncthreads();
  }

  const int mW = mBlock + wm, fW = fBlock + wf;
#pragma unroll
  for (int i = 0; i < 4; ++i)
#pragma unroll
    for (int j = 0; j < 4; ++j) {
      const int f = fW + j * 16 + ln;
      const float bia = bias[f];
#pragma unroll
      for (int r = 0; r < 8; ++r) {
        const int m = mW + i * 16 + r + hi8 * 8;
        out[(size_t)m * EMB + f] = acc[i][j][r] + bia;
      }
    }
}

// ---------------------------------------------------------------------------
extern "C" void kernel_launch(void* const* d_in, const int* in_sizes, int n_in,
                              void* d_out, int out_size, void* d_ws, size_t ws_size,
                              hipStream_t stream) {
  const float* x      = (const float*)d_in[0];
  const float* w_qkv  = (const float*)d_in[1];
  const float* b_qkv  = (const float*)d_in[2];
  const float* w_proj = (const float*)d_in[3];
  const float* b_proj = (const float*)d_in[4];
  float* out = (float*)d_out;

  char* ws = (char*)d_ws;
  const size_t SEG = (size_t)NB * NH * NSEQ * HD * sizeof(__bf16);  // 16 MB
  __bf16* qb   = (__bf16*)(ws);
  __bf16* kb   = (__bf16*)(ws + SEG);
  __bf16* vt   = (__bf16*)(ws + 2 * SEG);
  __bf16* attn = (__bf16*)(ws + 3 * SEG);

  // 1) QKV projection: M=8192, F=3072 -> grid 64 x 24.
  qkv_kernel<<<dim3(64, 24), 128, 0, stream>>>(x, w_qkv, b_qkv, qb, kb, vt);
  // 2) Flash attention: grid (N/128, B*H) = (8, 128).
  attn_kernel<<<dim3(8, 128), 128, 0, stream>>>(qb, kb, vt, attn);
  // 3) Output projection: M=8192, F=1024 -> grid 64 x 8.
  proj_kernel<<<dim3(64, 8), 128, 0, stream>>>(attn, w_proj, b_proj, out);
}
